// MatGPTQLinear_90718299226265
// MI455X (gfx1250) — compile-verified, matrix operand order checked by reading, not run on hardware
//
#include <hip/hip_runtime.h>

typedef __attribute__((ext_vector_type(16))) _Float16 v16h;
typedef __attribute__((ext_vector_type(8)))  _Float16 v8h;
typedef __attribute__((ext_vector_type(8)))  float    v8f;
typedef __attribute__((ext_vector_type(4)))  float    v4f;
typedef __attribute__((ext_vector_type(4)))  int      v4i;

#define TOKENS 8192
#define N_IN   4096
#define N_OUT  4096
#define GS     128
#define NGROUP (N_IN / GS)   // 32

constexpr int BM = 128;
constexpr int BN = 128;
constexpr int BK = 128;              // == GROUP_SIZE
constexpr int LDT = BK + 8;          // LDS row stride (f16): 272B/row, 16B aligned
constexpr unsigned TILE_H = BM * LDT;     // halves per tile (17408)
constexpr unsigned TILE_B = TILE_H * 2;   // 34816 bytes
constexpr unsigned BUF_B  = TILE_B * 2;   // X + W tile per buffer (69632)
constexpr unsigned SMEM_B = BUF_B * 2;    // ping-pong: 139264 B -> 2 blocks/WGP

// ---------------------------------------------------------------------------
// 16x32 f16 WMMA fragment from LDS (ISA 7.12.2 16-bit A layout):
//  lanes 0-15 : row = lane,    elems 0-7 -> K 0-7,  8-15 -> K 16-23
//  lanes 16-31: row = lane-16, elems 0-7 -> K 8-15, 8-15 -> K 24-31
// ---------------------------------------------------------------------------
__device__ __forceinline__ v16h load_frag_lds(const _Float16* base, int row0,
                                              int k0, int lane)
{
  const int r    = row0 + (lane & 15);
  const int hsel = lane >> 4;
  const _Float16* p = base + r * LDT + k0 + hsel * 8;
  v8h lo = *(const v8h*)(p);        // ds_load_b128
  v8h hi = *(const v8h*)(p + 16);   // ds_load_b128
  v16h f;
#pragma unroll
  for (int i = 0; i < 8; ++i) { f[i] = lo[i]; f[i + 8] = hi[i]; }
  return f;
}

__device__ __forceinline__ v8f wmma_f16(v16h a, v16h b, v8f c)
{
  return __builtin_amdgcn_wmma_f32_16x16x32_f16(false, a, false, b,
                                                (short)0, c, false, false);
}

// ===========================================================================
// Phase-A kernels (workspace path)
// ===========================================================================

// W[o,i] = (q-8)*scale  ->  f16, once. 8 elems/thread, 16B stores.
__global__ __launch_bounds__(256)
void dequant_w_kernel(const int* __restrict__ qw, const float* __restrict__ sc,
                      _Float16* __restrict__ wh)
{
  unsigned idx = blockIdx.x * 256u + threadIdx.x;     // over N_OUT*N_IN/8
  unsigned row = idx >> 9;                            // N_IN/8 = 512 chunks/row
  unsigned c8  = (idx & 511u) << 3;
  float s = sc[(size_t)row * NGROUP + (c8 >> 7)];     // GS = 128
  const int* p = qw + (size_t)row * N_IN + c8;
  v4i q0 = *(const v4i*)(p);
  v4i q1 = *(const v4i*)(p + 4);
  v8h o;
  o[0] = (_Float16)((float)(q0.x - 8) * s);
  o[1] = (_Float16)((float)(q0.y - 8) * s);
  o[2] = (_Float16)((float)(q0.z - 8) * s);
  o[3] = (_Float16)((float)(q0.w - 8) * s);
  o[4] = (_Float16)((float)(q1.x - 8) * s);
  o[5] = (_Float16)((float)(q1.y - 8) * s);
  o[6] = (_Float16)((float)(q1.z - 8) * s);
  o[7] = (_Float16)((float)(q1.w - 8) * s);
  *(v8h*)(wh + (size_t)row * N_IN + c8) = o;
}

// x fp32 -> f16, once.
__global__ __launch_bounds__(256)
void cvt_x_kernel(const float* __restrict__ x, _Float16* __restrict__ xh)
{
  unsigned idx = blockIdx.x * 256u + threadIdx.x;     // over TOKENS*N_IN/8
  unsigned row = idx >> 9;
  unsigned c8  = (idx & 511u) << 3;
  const float* p = x + (size_t)row * N_IN + c8;
  v4f a = *(const v4f*)(p);
  v4f b = *(const v4f*)(p + 4);
  v8h o;
  o[0] = (_Float16)a.x; o[1] = (_Float16)a.y; o[2] = (_Float16)a.z; o[3] = (_Float16)a.w;
  o[4] = (_Float16)b.x; o[5] = (_Float16)b.y; o[6] = (_Float16)b.z; o[7] = (_Float16)b.w;
  *(v8h*)(xh + (size_t)row * N_IN + c8) = o;
}

// f16 GEMM: 4 waves, 64x64 per wave (16 WMMA / 8 frag-loads per K-step),
// ping-pong LDS fed by CDNA5 async global->LDS copies.
__global__ __launch_bounds__(128)
void gemm_f16_async_kernel(const _Float16* __restrict__ xh,
                           const _Float16* __restrict__ wh,
                           float* __restrict__ out)
{
  extern __shared__ v8h smem_raw[];                 // 16B-aligned dynamic LDS
  _Float16* smem = (_Float16*)smem_raw;

  const int tid  = threadIdx.x;
  const int lane = tid & 31;
  const int wave = tid >> 5;     // 0..3
  const int wm   = wave & 1;     // 2 waves across M
  const int wn   = wave >> 1;    // 2 waves across N
  const int m0   = blockIdx.x * BM;
  const int n0   = blockIdx.y * BN;

  const unsigned smem_base = (unsigned)(size_t)(void*)smem;

  v8f acc[4][4];
#pragma unroll
  for (int i = 0; i < 4; ++i)
#pragma unroll
    for (int j = 0; j < 4; ++j)
      acc[i][j] = (v8f)0.0f;

  // Issue one k-tile's async copies: 16 x-chunks + 16 w-chunks of 16B / thread.
  auto stage_async = [&](unsigned bufbyte, const _Float16* gx, const _Float16* gw) {
#pragma unroll
    for (int it = 0; it < 16; ++it) {
      int idx = it * 128 + tid;            // 2048 16B chunks per tile
      int row = idx >> 4;                  // 16 chunks per row
      int c8  = (idx & 15) << 3;
      unsigned lx = smem_base + bufbyte + (unsigned)(row * LDT + c8) * 2u;
      unsigned lw = lx + TILE_B;
      const void* px = gx + (size_t)row * N_IN + c8;
      const void* pw = gw + (size_t)row * N_IN + c8;
      asm volatile("global_load_async_to_lds_b128 %0, %1, off"
                   :: "v"(lx), "v"(px) : "memory");
      asm volatile("global_load_async_to_lds_b128 %0, %1, off"
                   :: "v"(lw), "v"(pw) : "memory");
    }
  };

  const _Float16* xg = xh + (size_t)m0 * N_IN;
  const _Float16* wg = wh + (size_t)n0 * N_IN;

  stage_async(0u, xg, wg);                            // 32 async ops in flight

  const int KT = N_IN / BK;                           // 32
  for (int kb = 0; kb < KT; ++kb) {
    const int cur = kb & 1;

    if (kb + 1 < KT) {
      stage_async((unsigned)((cur ^ 1) * BUF_B), xg + (kb + 1) * BK,
                  wg + (kb + 1) * BK);                // +32 newer async ops
      // current tile's 32 older ops retired when <=32 remain outstanding
      asm volatile("s_wait_asynccnt 0x20" ::: "memory");
    } else {
      asm volatile("s_wait_asynccnt 0x0" ::: "memory");
    }
    __syncthreads();                                  // all waves' data visible

    const _Float16* lX = smem + (unsigned)cur * (BUF_B / 2);
    const _Float16* lW = lX + TILE_H;

#pragma unroll
    for (int ks = 0; ks < BK; ks += 32) {
      v16h a0 = load_frag_lds(lX, wm * 64 +  0, ks, lane);
      v16h a1 = load_frag_lds(lX, wm * 64 + 16, ks, lane);
      v16h a2 = load_frag_lds(lX, wm * 64 + 32, ks, lane);
      v16h a3 = load_frag_lds(lX, wm * 64 + 48, ks, lane);
      v16h b0 = load_frag_lds(lW, wn * 64 +  0, ks, lane);
      v16h b1 = load_frag_lds(lW, wn * 64 + 16, ks, lane);
      v16h b2 = load_frag_lds(lW, wn * 64 + 32, ks, lane);
      v16h b3 = load_frag_lds(lW, wn * 64 + 48, ks, lane);
      acc[0][0] = wmma_f16(a0, b0, acc[0][0]);
      acc[0][1] = wmma_f16(a0, b1, acc[0][1]);
      acc[0][2] = wmma_f16(a0, b2, acc[0][2]);
      acc[0][3] = wmma_f16(a0, b3, acc[0][3]);
      acc[1][0] = wmma_f16(a1, b0, acc[1][0]);
      acc[1][1] = wmma_f16(a1, b1, acc[1][1]);
      acc[1][2] = wmma_f16(a1, b2, acc[1][2]);
      acc[1][3] = wmma_f16(a1, b3, acc[1][3]);
      acc[2][0] = wmma_f16(a2, b0, acc[2][0]);
      acc[2][1] = wmma_f16(a2, b1, acc[2][1]);
      acc[2][2] = wmma_f16(a2, b2, acc[2][2]);
      acc[2][3] = wmma_f16(a2, b3, acc[2][3]);
      acc[3][0] = wmma_f16(a3, b0, acc[3][0]);
      acc[3][1] = wmma_f16(a3, b1, acc[3][1]);
      acc[3][2] = wmma_f16(a3, b2, acc[3][2]);
      acc[3][3] = wmma_f16(a3, b3, acc[3][3]);
    }
    __syncthreads();                                  // done reading cur buffer
  }

  // Epilogue: C layout — VGPR v -> M = v + 8*(lane>=16), N = lane&15
  const int hsel = lane >> 4;
  const int cn   = lane & 15;
#pragma unroll
  for (int mi = 0; mi < 4; ++mi)
#pragma unroll
    for (int ni = 0; ni < 4; ++ni)
#pragma unroll
      for (int v = 0; v < 8; ++v) {
        int row = m0 + wm * 64 + mi * 16 + hsel * 8 + v;
        int col = n0 + wn * 64 + ni * 16 + cn;
        out[(size_t)row * N_OUT + col] = acc[mi][ni][v];
      }
}

// ===========================================================================
// Fallback: fused dequant+GEMM (no workspace needed)
// ===========================================================================
__global__ __launch_bounds__(256)
void gptq_fused_kernel(const float* __restrict__ x,
                       const int*   __restrict__ qw,
                       const float* __restrict__ scales,
                       float* __restrict__ out)
{
  __shared__ __align__(16) _Float16 ldsX[BM * LDT];
  __shared__ __align__(16) _Float16 ldsW[BN * LDT];

  const int tid  = threadIdx.x;
  const int lane = tid & 31;
  const int wave = tid >> 5;
  const int wm   = wave & 1;
  const int wn   = wave >> 1;
  const int m0 = blockIdx.x * BM;
  const int n0 = blockIdx.y * BN;

  v8f acc[4][2];
#pragma unroll
  for (int i = 0; i < 4; ++i)
#pragma unroll
    for (int j = 0; j < 2; ++j)
      acc[i][j] = (v8f)0.0f;

  for (int kb = 0; kb < N_IN / BK; ++kb) {
    const int k0 = kb * BK;

#pragma unroll
    for (int it = 0; it < 16; ++it) {
      int idx = it * 256 + tid;
      int row = idx >> 5;
      int c4  = (idx & 31) << 2;
      v4f v = *(const v4f*)(x + (size_t)(m0 + row) * N_IN + k0 + c4);
      _Float16* d = ldsX + row * LDT + c4;
      d[0] = (_Float16)v.x; d[1] = (_Float16)v.y;
      d[2] = (_Float16)v.z; d[3] = (_Float16)v.w;
    }
#pragma unroll
    for (int it = 0; it < 16; ++it) {
      int idx = it * 256 + tid;
      int row = idx >> 5;
      int c4  = (idx & 31) << 2;
      v4i q = *(const v4i*)(qw + (size_t)(n0 + row) * N_IN + k0 + c4);
      float s = scales[(size_t)(n0 + row) * NGROUP + kb];
      _Float16* d = ldsW + row * LDT + c4;
      d[0] = (_Float16)((float)(q.x - 8) * s);
      d[1] = (_Float16)((float)(q.y - 8) * s);
      d[2] = (_Float16)((float)(q.z - 8) * s);
      d[3] = (_Float16)((float)(q.w - 8) * s);
    }
    if (kb + 1 < N_IN / BK) {
      __builtin_prefetch(x  + (size_t)(m0 + (tid >> 1)) * N_IN + k0 + BK + (tid & 1) * 64, 0, 3);
      __builtin_prefetch(qw + (size_t)(n0 + (tid >> 1)) * N_IN + k0 + BK + (tid & 1) * 64, 0, 3);
    }
    __syncthreads();

#pragma unroll
    for (int ks = 0; ks < BK; ks += 32) {
      v16h b0 = load_frag_lds(ldsW, wn * 32 +  0, ks, lane);
      v16h b1 = load_frag_lds(ldsW, wn * 32 + 16, ks, lane);
      v16h a0 = load_frag_lds(ldsX, wm * 64 +  0, ks, lane);
      v16h a1 = load_frag_lds(ldsX, wm * 64 + 16, ks, lane);
      v16h a2 = load_frag_lds(ldsX, wm * 64 + 32, ks, lane);
      v16h a3 = load_frag_lds(ldsX, wm * 64 + 48, ks, lane);
      acc[0][0] = wmma_f16(a0, b0, acc[0][0]);
      acc[0][1] = wmma_f16(a0, b1, acc[0][1]);
      acc[1][0] = wmma_f16(a1, b0, acc[1][0]);
      acc[1][1] = wmma_f16(a1, b1, acc[1][1]);
      acc[2][0] = wmma_f16(a2, b0, acc[2][0]);
      acc[2][1] = wmma_f16(a2, b1, acc[2][1]);
      acc[3][0] = wmma_f16(a3, b0, acc[3][0]);
      acc[3][1] = wmma_f16(a3, b1, acc[3][1]);
    }
    __syncthreads();
  }

  const int hsel = lane >> 4;
  const int cn   = lane & 15;
#pragma unroll
  for (int mi = 0; mi < 4; ++mi)
#pragma unroll
    for (int ni = 0; ni < 2; ++ni)
#pragma unroll
      for (int v = 0; v < 8; ++v) {
        int row = m0 + wm * 64 + mi * 16 + hsel * 8 + v;
        int col = n0 + wn * 32 + ni * 16 + cn;
        out[(size_t)row * N_OUT + col] = acc[mi][ni][v];
      }
}

// ===========================================================================
extern "C" void kernel_launch(void* const* d_in, const int* in_sizes, int n_in,
                              void* d_out, int out_size, void* d_ws, size_t ws_size,
                              hipStream_t stream) {
  const float* x  = (const float*)d_in[0];   // [TOKENS, N_IN] fp32
  const int*   qw = (const int*)d_in[1];     // [N_OUT, N_IN] int32 (0..15)
  const float* sc = (const float*)d_in[2];   // [N_OUT, N_IN/GS] fp32
  float* out = (float*)d_out;                // [TOKENS, N_OUT] fp32

  const size_t WB = (size_t)N_OUT * N_IN * sizeof(_Float16);   // 32 MB
  const size_t XB = (size_t)TOKENS * N_IN * sizeof(_Float16);  // 64 MB

  dim3 grid(TOKENS / BM, N_OUT / BN);        // (64, 32)

  if (ws_size >= WB + XB) {
    _Float16* wh = (_Float16*)d_ws;
    _Float16* xh = (_Float16*)((char*)d_ws + WB);
    dequant_w_kernel<<<(unsigned)((size_t)N_OUT * N_IN / 8 / 256), 256, 0, stream>>>(qw, sc, wh);
    cvt_x_kernel<<<(unsigned)((size_t)TOKENS * N_IN / 8 / 256), 256, 0, stream>>>(x, xh);
    gemm_f16_async_kernel<<<grid, 128, SMEM_B, stream>>>(xh, wh, out);
  } else {
    gptq_fused_kernel<<<grid, 256, 0, stream>>>(x, qw, sc, out);
  }
}
